// WaveNet_14542759264852
// MI455X (gfx1250) — matrix operand order repaired and checked by reading, not compiled, not run on hardware
//
#include <hip/hip_runtime.h>
#include <hip/hip_bf16.h>
#include <math.h>

#define CHN 32
#define LEN 8192
#define BSZ 16
#define NBLK 24

typedef __attribute__((ext_vector_type(16))) __bf16 v16bf;
typedef __attribute__((ext_vector_type(8)))  float  v8f;

static __device__ __forceinline__ v8f wmma_bf16(v16bf a, v16bf b, v8f c) {
  // D = A(16x32 bf16) x B(32x16 bf16) + C(16x16 f32)
  return __builtin_amdgcn_wmma_f32_16x16x32_bf16(false, a, false, b, (short)0, c,
                                                 false, false);
}

// ---------------------------------------------------------------------------
// Kernel 1: initial causal conv (1 -> 32 channels, K=2, dil=1).
// Writes running h (buffer A) and the saved h0 copy (skip_sum = h_final - h0).
// h layout: (B, L, 32) float, time-major so channel vectors are contiguous.
// ---------------------------------------------------------------------------
__global__ __launch_bounds__(256) void k_init(const float* __restrict__ x,
                                              const float* __restrict__ W0,
                                              const float* __restrict__ b0,
                                              float* __restrict__ hA,
                                              float* __restrict__ h0) {
  const int gid = blockIdx.x * 256 + threadIdx.x;   // (b,t) flat
  const int b = gid >> 13;
  const int t = gid & (LEN - 1);
  const float xc = x[b * LEN + t];
  const float xp = (t > 0) ? x[b * LEN + t - 1] : 0.f;
  const size_t base = (size_t)gid * CHN;
#pragma unroll
  for (int o = 0; o < CHN; ++o) {
    float v = W0[o * 2 + 0] * xp + W0[o * 2 + 1] * xc + b0[o];
    hA[base + o] = v;
    h0[base + o] = v;
  }
}

// ---------------------------------------------------------------------------
// Kernel 2: one WaveNet residual block, all GEMMs on v_wmma_f32_16x16x32_bf16.
// Grid: 16 batches x 16 chunks (512 t each); 8 waves/block, 4 16-wide tiles
// per wave. Reads hin (layer input), writes hout = hin + skip (ping-pong).
// ---------------------------------------------------------------------------
__global__ __launch_bounds__(256) void k_layer(const float* __restrict__ hin,
                                               float* __restrict__ hout,
                                               const float* __restrict__ Wd,
                                               const float* __restrict__ bd,
                                               const float* __restrict__ W1,
                                               const float* __restrict__ b1,
                                               int dil) {
  const int lane = threadIdx.x & 31;
  const int wave = threadIdx.x >> 5;
  const int tl = lane & 15;   // column (time within tile)
  const int hh = lane >> 4;   // lane half
  const int b = blockIdx.x >> 4;
  const int chunkBase = (blockIdx.x & 15) * (LEN / 16);

  // A-fragment K (=input channel) map (contiguous k order; B fragments use the same):
  // lanes 0-15: K = {0..15}; lanes 16-31: K = {16..31}.
  int kmap[16];
#pragma unroll
  for (int j = 0; j < 16; ++j) kmap[j] = 16 * hh + j;

  // Conv weights: Wd is (64, 32, 2); tap k applied to h[t - (1-k)*dil].
  v16bf Ac0[4], Ac1[4];
#pragma unroll
  for (int m = 0; m < 4; ++m) {
    v16bf a0, a1;
#pragma unroll
    for (int j = 0; j < 16; ++j) {
      const int o = 16 * m + tl;  // output channel = A row
      const float* wp = Wd + (o * CHN + kmap[j]) * 2;
      a0[j] = (__bf16)wp[0];
      a1[j] = (__bf16)wp[1];
    }
    Ac0[m] = a0;
    Ac1[m] = a1;
  }
  // Skip 1x1 weights: W1 is (32, 32).
  v16bf A1f[2];
#pragma unroll
  for (int m = 0; m < 2; ++m) {
    v16bf a;
#pragma unroll
    for (int j = 0; j < 16; ++j) a[j] = (__bf16)W1[(16 * m + tl) * CHN + kmap[j]];
    A1f[m] = a;
  }
  // Bias fragments in C/D layout (row = 16m + r + 8*hh).
  float zb[4][8], s1b[2][8];
#pragma unroll
  for (int m = 0; m < 4; ++m)
#pragma unroll
    for (int r = 0; r < 8; ++r) zb[m][r] = bd[16 * m + 8 * hh + r];
#pragma unroll
  for (int m = 0; m < 2; ++m)
#pragma unroll
    for (int r = 0; r < 8; ++r) s1b[m][r] = b1[16 * m + 8 * hh + r];

  const size_t bBase = (size_t)b * LEN * CHN;

  for (int tile = 0; tile < 4; ++tile) {
    const int t0 = chunkBase + (wave * 4 + tile) * 16;
    const int tcur = t0 + tl;

    // B fragment (current tap): lane holds channels 16*hh..16*hh+15 at tcur.
    const float* pc = hin + bBase + (size_t)tcur * CHN + 16 * hh;
    const float4 c0 = ((const float4*)pc)[0];
    const float4 c1 = ((const float4*)pc)[1];
    const float4 c2 = ((const float4*)pc)[2];
    const float4 c3 = ((const float4*)pc)[3];
    v16bf Bcur;
    Bcur[0] = (__bf16)c0.x; Bcur[1] = (__bf16)c0.y; Bcur[2] = (__bf16)c0.z; Bcur[3] = (__bf16)c0.w;
    Bcur[4] = (__bf16)c1.x; Bcur[5] = (__bf16)c1.y; Bcur[6] = (__bf16)c1.z; Bcur[7] = (__bf16)c1.w;
    Bcur[8] = (__bf16)c2.x; Bcur[9] = (__bf16)c2.y; Bcur[10] = (__bf16)c2.z; Bcur[11] = (__bf16)c2.w;
    Bcur[12] = (__bf16)c3.x; Bcur[13] = (__bf16)c3.y; Bcur[14] = (__bf16)c3.z; Bcur[15] = (__bf16)c3.w;

    // B fragment (dilated tap, causal zero-pad).
    v16bf Bprev;
    const int tprev = tcur - dil;
    if (tprev >= 0) {
      const float* pp = hin + bBase + (size_t)tprev * CHN + 16 * hh;
      const float4 p0 = ((const float4*)pp)[0];
      const float4 p1 = ((const float4*)pp)[1];
      const float4 p2 = ((const float4*)pp)[2];
      const float4 p3 = ((const float4*)pp)[3];
      Bprev[0] = (__bf16)p0.x; Bprev[1] = (__bf16)p0.y; Bprev[2] = (__bf16)p0.z; Bprev[3] = (__bf16)p0.w;
      Bprev[4] = (__bf16)p1.x; Bprev[5] = (__bf16)p1.y; Bprev[6] = (__bf16)p1.z; Bprev[7] = (__bf16)p1.w;
      Bprev[8] = (__bf16)p2.x; Bprev[9] = (__bf16)p2.y; Bprev[10] = (__bf16)p2.z; Bprev[11] = (__bf16)p2.w;
      Bprev[12] = (__bf16)p3.x; Bprev[13] = (__bf16)p3.y; Bprev[14] = (__bf16)p3.z; Bprev[15] = (__bf16)p3.w;
    } else {
#pragma unroll
      for (int j = 0; j < 16; ++j) Bprev[j] = (__bf16)0.f;
    }

    // Z(64x16) = Wtap0 * H(t-dil) + Wtap1 * H(t) + bd   -> 8 WMMAs
    v8f accZ[4];
#pragma unroll
    for (int m = 0; m < 4; ++m) {
#pragma unroll
      for (int r = 0; r < 8; ++r) accZ[m][r] = zb[m][r];
      accZ[m] = wmma_bf16(Ac0[m], Bprev, accZ[m]);
      accZ[m] = wmma_bf16(Ac1[m], Bcur, accZ[m]);
    }

    // Gated activation: channels line up per-lane (f = m, g = m+2).
    float gf0[8], gf1[8];
#pragma unroll
    for (int r = 0; r < 8; ++r) {
      gf0[r] = tanhf(accZ[0][r]) * (1.f / (1.f + __expf(-accZ[2][r])));
      gf1[r] = tanhf(accZ[1][r]) * (1.f / (1.f + __expf(-accZ[3][r])));
    }

    // D-layout -> B-layout: swap one fragment between lane halves (same time
    // column lives in lane l and l+16), then pack to bf16.
    v16bf Bg;
#pragma unroll
    for (int r = 0; r < 8; ++r) {
      float send = hh ? gf0[r] : gf1[r];
      float recv = __shfl_xor(send, 16, 32);
      float lo = hh ? recv : gf0[r];
      float hi = hh ? gf1[r] : recv;
      Bg[r] = (__bf16)lo;
      Bg[8 + r] = (__bf16)hi;
    }

    // skip(32x16) = W1 * gated + b1   -> 2 WMMAs
    v8f s0, s1;
#pragma unroll
    for (int r = 0; r < 8; ++r) { s0[r] = s1b[0][r]; s1[r] = s1b[1][r]; }
    s0 = wmma_bf16(A1f[0], Bg, s0);
    s1 = wmma_bf16(A1f[1], Bg, s1);

    // Same half-swap to reach the coalesced writer layout, then
    // hout = hin + skip (residual); skip_sum recovered later as h_final-h0.
    float lo[8], hi[8];
#pragma unroll
    for (int r = 0; r < 8; ++r) {
      float send = hh ? s0[r] : s1[r];
      float recv = __shfl_xor(send, 16, 32);
      lo[r] = hh ? recv : s0[r];
      hi[r] = hh ? s1[r] : recv;
    }
    float* po = hout + bBase + (size_t)tcur * CHN + 16 * hh;
    ((float4*)po)[0] = make_float4(c0.x + lo[0], c0.y + lo[1], c0.z + lo[2], c0.w + lo[3]);
    ((float4*)po)[1] = make_float4(c1.x + lo[4], c1.y + lo[5], c1.z + lo[6], c1.w + lo[7]);
    ((float4*)po)[2] = make_float4(c2.x + hi[0], c2.y + hi[1], c2.z + hi[2], c2.w + hi[3]);
    ((float4*)po)[3] = make_float4(c3.x + hi[4], c3.y + hi[5], c3.z + hi[6], c3.w + hi[7]);
  }
}

// ---------------------------------------------------------------------------
// Kernel 3: head part 1. y = relu(Wf*relu(hF-h0)+bf), adaptive-max-pool(4).
// One block per (b, quarter); LDS tree reduction for per-channel max.
// ---------------------------------------------------------------------------
__global__ __launch_bounds__(256) void k_head1(const float* __restrict__ hF,
                                               const float* __restrict__ h0,
                                               const float* __restrict__ Wf,
                                               const float* __restrict__ bfv,
                                               float* __restrict__ pool) {
  __shared__ float sWf[CHN * CHN];
  __shared__ float sbf[CHN];
  __shared__ float red[256 * 33];
  const int tid = threadIdx.x;
  const int b = blockIdx.x >> 2;
  const int q = blockIdx.x & 3;
  for (int i = tid; i < CHN * CHN; i += 256) sWf[i] = Wf[i];
  if (tid < CHN) sbf[tid] = bfv[tid];
  __syncthreads();

  float m[CHN];
#pragma unroll
  for (int o = 0; o < CHN; ++o) m[o] = 0.f;  // relu output >= 0

  const int tBase = q * (LEN / 4);
  for (int it = 0; it < (LEN / 4) / 256; ++it) {
    const int t = tBase + it * 256 + tid;
    const size_t idx = ((size_t)b * LEN + t) * CHN;
    float v[CHN];
#pragma unroll
    for (int c = 0; c < CHN; ++c) {
      float d = hF[idx + c] - h0[idx + c];  // skip_sum
      v[c] = d > 0.f ? d : 0.f;
    }
#pragma unroll
    for (int o = 0; o < CHN; ++o) {
      float acc = sbf[o];
#pragma unroll
      for (int c = 0; c < CHN; ++c) acc = fmaf(sWf[o * CHN + c], v[c], acc);
      acc = acc > 0.f ? acc : 0.f;
      m[o] = fmaxf(m[o], acc);
    }
  }
#pragma unroll
  for (int o = 0; o < CHN; ++o) red[tid * 33 + o] = m[o];
  __syncthreads();
  for (int s = 128; s >= 1; s >>= 1) {
    if (tid < s) {
#pragma unroll
      for (int o = 0; o < CHN; ++o)
        red[tid * 33 + o] = fmaxf(red[tid * 33 + o], red[(tid + s) * 33 + o]);
    }
    __syncthreads();
  }
  if (tid < CHN) pool[b * (CHN * 4) + tid * 4 + q] = red[tid];  // (B, C, 4) -> flat c*4+q
}

// ---------------------------------------------------------------------------
// Kernel 4: tiny FC head. 128->64 relu, 64->10. One block.
// ---------------------------------------------------------------------------
__global__ __launch_bounds__(256) void k_head2(const float* __restrict__ pool,
                                               const float* __restrict__ fW1,
                                               const float* __restrict__ fb1,
                                               const float* __restrict__ fW2,
                                               const float* __restrict__ fb2,
                                               float* __restrict__ out) {
  __shared__ float y1[BSZ * 64];
  const int tid = threadIdx.x;
  for (int e = tid; e < BSZ * 64; e += 256) {
    const int b = e >> 6, j = e & 63;
    float acc = fb1[j];
    for (int k = 0; k < 128; ++k) acc = fmaf(fW1[j * 128 + k], pool[b * 128 + k], acc);
    y1[e] = acc > 0.f ? acc : 0.f;
  }
  __syncthreads();
  if (tid < BSZ * 10) {
    const int b = tid / 10, i = tid % 10;
    float acc = fb2[i];
    for (int j = 0; j < 64; ++j) acc = fmaf(fW2[i * 64 + j], y1[b * 64 + j], acc);
    out[tid] = acc;
  }
}

extern "C" void kernel_launch(void* const* d_in, const int* in_sizes, int n_in,
                              void* d_out, int out_size, void* d_ws, size_t ws_size,
                              hipStream_t stream) {
  (void)in_sizes; (void)n_in; (void)out_size; (void)ws_size;
  const float* x   = (const float*)d_in[0];
  const float* W0  = (const float*)d_in[1];
  const float* b0  = (const float*)d_in[2];
  const float* Wd  = (const float*)d_in[3];
  const float* bd  = (const float*)d_in[4];
  const float* W1  = (const float*)d_in[5];
  const float* b1  = (const float*)d_in[6];
  const float* Wf  = (const float*)d_in[7];
  const float* bf  = (const float*)d_in[8];
  const float* fW1 = (const float*)d_in[9];
  const float* fb1 = (const float*)d_in[10];
  const float* fW2 = (const float*)d_in[11];
  const float* fb2 = (const float*)d_in[12];
  float* out = (float*)d_out;

  float* w = (float*)d_ws;
  float* pool = w;                                   // 2048 floats
  float* hA = w + 4096;                              // (B,L,32)
  float* hB = hA + (size_t)BSZ * LEN * CHN;          // ping-pong buffer
  float* h0 = hB + (size_t)BSZ * LEN * CHN;          // saved initial h

  k_init<<<BSZ * LEN / 256, 256, 0, stream>>>(x, W0, b0, hA, h0);

  for (int i = 0; i < NBLK; ++i) {
    const int dil = 1 << (i & 7);
    const float* hin = (i & 1) ? hB : hA;
    float* hout = (i & 1) ? hA : hB;
    k_layer<<<BSZ * 16, 256, 0, stream>>>(hin, hout,
                                          Wd + (size_t)i * (2 * CHN * CHN * 2),
                                          bd + i * (2 * CHN),
                                          W1 + i * (CHN * CHN),
                                          b1 + i * CHN, dil);
  }
  // 24 layers (even) -> final h lands back in hA.
  k_head1<<<BSZ * 4, 256, 0, stream>>>(hA, h0, Wf, bf, pool);
  k_head2<<<1, 256, 0, stream>>>(pool, fW1, fb1, fW2, fb2, out);
}
